// GCN_12481174962469
// MI455X (gfx1250) — matrix operand order repaired and checked by reading, not compiled, not run on hardware
//
#include <hip/hip_runtime.h>

// ---------------------------------------------------------------------------
// GCN forward for MI455X (gfx1250, wave32, WMMA).
//   x   = relu(NF @ W_emb + b_emb)            (fused GEMM #1, bf16 WMMA)
//   xw  = x @ W_conv                          (fused GEMM #2, same kernel)
//   deg = 1 + indeg(tgt); dinv = rsqrt(deg)
//   acc = b_conv + dinv^2 * xw                (self loop)  + edge scatter
//   pooled mean over batch; out = pooled @ W_lin + b_lin
// ---------------------------------------------------------------------------

#define Nn 100000
#define Ee 1600000
#define Hh 128
#define Gg 64
#define LDP 136   // padded LDS leading dim (elements): 272 B pitch, no bank conflicts

typedef __attribute__((ext_vector_type(16))) __bf16 v16bf;
typedef __attribute__((ext_vector_type(8)))  __bf16 v8bf;
typedef __attribute__((ext_vector_type(8)))  float  v8f;

// ---- WMMA fragment loads, 2 x ds_load_b128 each (CDNA5 ISA 7.12.2) ---------
// A (16x32 bf16) from row-major LDS tile: lane r=lane&15 holds row m0+r.
// Slot s in 0..7  -> k = k0 + hi*8 + s        (contiguous 16B)
// Slot s in 8..15 -> k = k0 + 16 + hi*8 + s-8 (contiguous 16B)
__device__ inline v16bf afrag(const __bf16* t, int m0, int k0, int lane) {
    int r = lane & 15, hi = lane >> 4;
    const __bf16* p = t + (m0 + r) * LDP + k0 + hi * 8;
    v8bf lo = *(const v8bf*)(p);
    v8bf hh = *(const v8bf*)(p + 16);
    return __builtin_shufflevector(lo, hh, 0, 1, 2, 3, 4, 5, 6, 7,
                                           8, 9, 10, 11, 12, 13, 14, 15);
}
// B (32x16 bf16) from TRANSPOSED LDS weights Wt[n*LDP + k]: lane c=lane&15
// holds column n0+c; slots s=0..7 -> k = k0 + hi*8 + s, s=8..15 -> k = k0 + 16 + hi*8 + s-8.
__device__ inline v16bf bfrag(const __bf16* Wt, int k0, int n0, int lane) {
    int c = lane & 15, hi = lane >> 4;
    const __bf16* p = Wt + (n0 + c) * LDP + k0 + hi * 8;
    v8bf lo = *(const v8bf*)(p);
    v8bf hh = *(const v8bf*)(p + 16);
    return __builtin_shufflevector(lo, hh, 0, 1, 2, 3, 4, 5, 6, 7,
                                           8, 9, 10, 11, 12, 13, 14, 15);
}

// ---- Kernel 1: fused double GEMM with WMMA ---------------------------------
__global__ __launch_bounds__(256) void k_gemm_fused(
    const float* __restrict__ nf, const float* __restrict__ Wemb,
    const float* __restrict__ bemb, const float* __restrict__ Wconv,
    float* __restrict__ xw)
{
    __shared__ __bf16 sWe[Hh * LDP];   // W_emb^T  (bf16, padded)  ~34.8 KB
    __shared__ __bf16 sWc[Hh * LDP];   // W_conv^T                 ~34.8 KB
    __shared__ __bf16 sA [64 * LDP];   // node-feature tile        ~17.4 KB
    __shared__ __bf16 sX [64 * LDP];   // relu(x) tile             ~17.4 KB

    const int tid  = threadIdx.x;
    const int lane = tid & 31;
    const int wid  = tid >> 5;                  // 8 waves
    const long base = (long)blockIdx.x * 64;    // first row of this tile

    // Stage both weight matrices transposed as bf16: Wt[n*LDP + k] = W[k][n].
#pragma unroll 4
    for (int i = tid; i < Hh * Hh / 4; i += 256) {
        int k = (i * 4) / Hh, n = (i * 4) % Hh;
        float4 w = ((const float4*)Wemb)[i];
        sWe[(n + 0) * LDP + k] = (__bf16)w.x; sWe[(n + 1) * LDP + k] = (__bf16)w.y;
        sWe[(n + 2) * LDP + k] = (__bf16)w.z; sWe[(n + 3) * LDP + k] = (__bf16)w.w;
        float4 v = ((const float4*)Wconv)[i];
        sWc[(n + 0) * LDP + k] = (__bf16)v.x; sWc[(n + 1) * LDP + k] = (__bf16)v.y;
        sWc[(n + 2) * LDP + k] = (__bf16)v.z; sWc[(n + 3) * LDP + k] = (__bf16)v.w;
    }
    // Stage 64x128 node-feature tile (coalesced float4, zero-pad past N).
#pragma unroll 4
    for (int i = tid; i < 64 * Hh / 4; i += 256) {
        int rl = (i * 4) / Hh, cl = (i * 4) % Hh;
        long row = base + rl;
        float4 v = {0.f, 0.f, 0.f, 0.f};
        if (row < Nn) v = ((const float4*)(nf + row * Hh))[cl / 4];
        __bf16* d = sA + rl * LDP + cl;     // 4 consecutive bf16, 8B aligned
        d[0] = (__bf16)v.x; d[1] = (__bf16)v.y; d[2] = (__bf16)v.z; d[3] = (__bf16)v.w;
    }
    __syncthreads();

    const int mt = wid & 3;           // M-tile 0..3 (16 rows each)
    const int nb = (wid >> 2) * 4;    // first of 4 N-tiles
    const int n  = lane & 15;
    const int mb = (lane >> 4) * 8;   // C/D layout: hi lanes hold M+8

    // ---- GEMM 1: x = relu(NF @ W_emb + b_emb) ----
    v8f acc[4] = {};
#pragma unroll
    for (int kk = 0; kk < 4; ++kk) {
        v16bf a = afrag(sA, mt * 16, kk * 32, lane);
#pragma unroll
        for (int nt = 0; nt < 4; ++nt) {
            v16bf b = bfrag(sWe, kk * 32, (nb + nt) * 16, lane);
            acc[nt] = __builtin_amdgcn_wmma_f32_16x16x32_bf16(
                false, a, false, b, (short)0, acc[nt], false, false);
        }
    }
#pragma unroll
    for (int nt = 0; nt < 4; ++nt) {
        int col = (nb + nt) * 16 + n;
        float bias = bemb[col];
#pragma unroll
        for (int v = 0; v < 8; ++v) {
            float val = acc[nt][v] + bias;
            val = val > 0.f ? val : 0.f;
            sX[(mt * 16 + mb + v) * LDP + col] = (__bf16)val;
        }
    }
    __syncthreads();

    // ---- GEMM 2: xw = x @ W_conv ----
    v8f z = {};
#pragma unroll
    for (int nt = 0; nt < 4; ++nt) acc[nt] = z;
#pragma unroll
    for (int kk = 0; kk < 4; ++kk) {
        v16bf a = afrag(sX, mt * 16, kk * 32, lane);
#pragma unroll
        for (int nt = 0; nt < 4; ++nt) {
            v16bf b = bfrag(sWc, kk * 32, (nb + nt) * 16, lane);
            acc[nt] = __builtin_amdgcn_wmma_f32_16x16x32_bf16(
                false, a, false, b, (short)0, acc[nt], false, false);
        }
    }
#pragma unroll
    for (int nt = 0; nt < 4; ++nt) {
        int col = (nb + nt) * 16 + n;
#pragma unroll
        for (int v = 0; v < 8; ++v) {
            long row = base + mt * 16 + mb + v;
            if (row < Nn) xw[row * Hh + col] = acc[nt][v];
        }
    }
}

// ---- tiny helper kernels ---------------------------------------------------
__global__ void k_fill(float* p, long n, float v) {
    long i = (long)blockIdx.x * blockDim.x + threadIdx.x;
    if (i < n) p[i] = v;
}
__global__ void k_deg(const int* __restrict__ ei, float* __restrict__ deg) {
    long e = (long)blockIdx.x * blockDim.x + threadIdx.x;
    if (e < Ee) atomicAdd(&deg[ei[Ee + e]], 1.0f);   // tgt row of edge_index
}
__global__ void k_rsqrt(float* d) {
    long i = (long)blockIdx.x * blockDim.x + threadIdx.x;
    if (i < Nn) d[i] = rsqrtf(d[i]);                 // deg >= 1 (self loop)
}
// acc = b_conv + dinv^2 * xw     (self-loop message + conv bias)
__global__ void k_agg_init(const float* __restrict__ xw, const float* __restrict__ dinv,
                           const float* __restrict__ bconv, float* __restrict__ acc) {
    long idx = (long)blockIdx.x * blockDim.x + threadIdx.x;   // one float4 each
    if (idx < (long)Nn * (Hh / 4)) {
        long i = idx >> 5;
        int  q = (int)(idx & 31) * 4;
        float d = dinv[i], nrm = d * d;
        float4 xv = *(const float4*)(xw + i * Hh + q);
        float4 bv = *(const float4*)(bconv + q);
        float4 r  = {bv.x + xv.x * nrm, bv.y + xv.y * nrm,
                     bv.z + xv.z * nrm, bv.w + xv.w * nrm};
        *(float4*)(acc + i * Hh + q) = r;
    }
}
// one wave32 per edge: gather xw[src] (float4/lane) and atomic-add into acc[tgt]
__global__ __launch_bounds__(256) void k_agg_edges(
    const int* __restrict__ ei, const float* __restrict__ dinv,
    const float* __restrict__ xw, float* __restrict__ acc) {
    const int lane = threadIdx.x & 31;
    long e = (long)blockIdx.x * 8 + (threadIdx.x >> 5);
    // prefetch a future edge's source row into cache (global_prefetch_b8)
    long ep = e + 256;
    if (ep < Ee) {
        int sp = ei[ep];
        __builtin_prefetch(xw + (long)sp * Hh + lane * 4, 0, 0);
    }
    if (e < Ee) {
        int s = ei[e], t = ei[Ee + e];
        float nrm = dinv[s] * dinv[t];
        float4 v = *(const float4*)(xw + (long)s * Hh + lane * 4);
        float* dst = acc + (long)t * Hh + lane * 4;
        atomicAdd(dst + 0, v.x * nrm);
        atomicAdd(dst + 1, v.y * nrm);
        atomicAdd(dst + 2, v.z * nrm);
        atomicAdd(dst + 3, v.w * nrm);
    }
}
// relu + mean-pool accumulation: one wave per node
__global__ __launch_bounds__(256) void k_pool(
    const float* __restrict__ acc, const int* __restrict__ batch,
    float* __restrict__ pooled, float* __restrict__ counts) {
    const int lane = threadIdx.x & 31;
    long i = (long)blockIdx.x * 8 + (threadIdx.x >> 5);
    if (i < Nn) {
        int g = batch[i];
        float4 v = *(const float4*)(acc + i * Hh + lane * 4);
        v.x = v.x > 0.f ? v.x : 0.f;  v.y = v.y > 0.f ? v.y : 0.f;
        v.z = v.z > 0.f ? v.z : 0.f;  v.w = v.w > 0.f ? v.w : 0.f;
        float* dst = pooled + (long)g * Hh + lane * 4;
        atomicAdd(dst + 0, v.x); atomicAdd(dst + 1, v.y);
        atomicAdd(dst + 2, v.z); atomicAdd(dst + 3, v.w);
        if (lane == 0) atomicAdd(&counts[g], 1.0f);
    }
}
__global__ void k_final(const float* __restrict__ pooled, const float* __restrict__ counts,
                        const float* __restrict__ Wlin, const float* __restrict__ blin,
                        float* __restrict__ out) {
    int g = threadIdx.x;
    if (g < Gg) {
        float c = counts[g]; c = c > 1.f ? c : 1.f;
        float s = 0.f;
#pragma unroll 4
        for (int h = 0; h < Hh; ++h) s += pooled[g * Hh + h] * Wlin[h];
        out[g] = s / c + blin[0];
    }
}

// ---------------------------------------------------------------------------
extern "C" void kernel_launch(void* const* d_in, const int* in_sizes, int n_in,
                              void* d_out, int out_size, void* d_ws, size_t ws_size,
                              hipStream_t stream) {
    const float* nf    = (const float*)d_in[0];
    const int*   ei    = (const int*)  d_in[2];   // [0:E)=src, [E:2E)=tgt
    const int*   batch = (const int*)  d_in[3];
    const float* Wemb  = (const float*)d_in[4];
    const float* bemb  = (const float*)d_in[5];
    const float* Wconv = (const float*)d_in[6];
    const float* bconv = (const float*)d_in[7];
    const float* Wlin  = (const float*)d_in[8];
    const float* blin  = (const float*)d_in[9];
    float* out = (float*)d_out;

    float* ws     = (float*)d_ws;                 // ~103 MB total
    float* xw     = ws;                           // N*H
    float* acc    = xw  + (long)Nn * Hh;          // N*H
    float* deg    = acc + (long)Nn * Hh;          // N   (becomes dinv in place)
    float* pooled = deg + Nn;                     // G*H
    float* counts = pooled + (long)Gg * Hh;       // G   (contiguous w/ pooled)

    // 1. fused WMMA GEMMs -> xw
    k_gemm_fused<<<(Nn + 63) / 64, 256, 0, stream>>>(nf, Wemb, bemb, Wconv, xw);
    // 2. degrees (self loop => start at 1), dinv = rsqrt(deg)
    k_fill<<<(Nn + 255) / 256, 256, 0, stream>>>(deg, Nn, 1.0f);
    k_deg<<<(Ee + 255) / 256, 256, 0, stream>>>(ei, deg);
    k_rsqrt<<<(Nn + 255) / 256, 256, 0, stream>>>(deg);
    // 3. init acc with bias + self-loop message, zero pooled/counts
    k_agg_init<<<((long)Nn * 32 + 255) / 256, 256, 0, stream>>>(xw, deg, bconv, acc);
    k_fill<<<(Gg * Hh + Gg + 255) / 256, 256, 0, stream>>>(pooled, Gg * Hh + Gg, 0.0f);
    // 4. edge scatter-add (L2-resident atomics)
    k_agg_edges<<<(Ee + 7) / 8, 256, 0, stream>>>(ei, deg, xw, acc);
    // 5. relu + mean pool
    k_pool<<<(Nn + 7) / 8, 256, 0, stream>>>(acc, batch, pooled, counts);
    // 6. final linear head
    k_final<<<1, 64, 0, stream>>>(pooled, counts, Wlin, blin, out);
}
